// middle_fssm_24292335026638
// MI455X (gfx1250) — matrix operand order repaired and checked by reading, not compiled, not run on hardware
//
#include <hip/hip_runtime.h>
#include <cmath>

// Problem sizes (fixed by the reference)
#define NSTATE 64
#define HCH    1024
#define LSEQ   2048
#define NBATCH 8

typedef __attribute__((ext_vector_type(16))) _Float16     v16h;
typedef __attribute__((ext_vector_type(8)))  float        v8f;
typedef __attribute__((ext_vector_type(2)))  float        v2f;
typedef __attribute__((ext_vector_type(4)))  unsigned int v4u;
typedef __attribute__((ext_vector_type(8)))  int          v8i;
typedef __attribute__((ext_vector_type(4)))  int          v4i;

// ---------------------------------------------------------------------------
// Kernel 0: per-frequency (h-independent) precompute.
//   z_l, g_l, Cauchy denominators cden[l][n] = 1/(g_l - Lam_n),
//   corr_l = k10/(1+k11), w2_l = 2/(1+z_l).
// ---------------------------------------------------------------------------
__global__ void k0_setup(const float* __restrict__ Br, const float* __restrict__ Bi,
                         const float* __restrict__ Pr, const float* __restrict__ Pi,
                         const float* __restrict__ Qr, const float* __restrict__ Qi,
                         const float* __restrict__ dre, const float* __restrict__ dim_,
                         const float* __restrict__ stepp,
                         float* __restrict__ cdenr, float* __restrict__ cdeni,
                         float* __restrict__ corr_r, float* __restrict__ corr_i,
                         float* __restrict__ w2r, float* __restrict__ w2i) {
  int l = blockIdx.x * blockDim.x + threadIdx.x;
  if (l >= LSEQ) return;
  float stepv = fmaxf(stepp[0], 1e-6f);
  float th = 6.28318530717958647692f * (float)l / (float)LSEQ;
  float zr = cosf(th), zi = -sinf(th);                 // z = exp(-2*pi*i*l/L)
  float ar = 1.f - zr, ai = -zi;                       // 1 - z
  float br2 = 1.f + zr, bi2 = zi;                      // 1 + z
  float inv_den = 1.f / (br2 * br2 + bi2 * bi2);
  float gs = 2.f / stepv;
  float gr = gs * (ar * br2 + ai * bi2) * inv_den;     // g = (2/step)(1-z)/(1+z)
  float gi = gs * (ai * br2 - ar * bi2) * inv_den;
  float k10r = 0.f, k10i = 0.f, k11r = 0.f, k11i = 0.f;
  for (int n = 0; n < NSTATE; ++n) {
    float dr = gr - dre[n], di = gi - dim_[n];
    float mg = 1.f / (dr * dr + di * di);
    float cr = dr * mg, ci = -di * mg;                 // 1/(g - Lam)
    cdenr[l * NSTATE + n] = cr;
    cdeni[l * NSTATE + n] = ci;
    float qbr = Qr[n] * Br[n] - Qi[n] * Bi[n];
    float qbi = Qr[n] * Bi[n] + Qi[n] * Br[n];
    float qpr = Qr[n] * Pr[n] - Qi[n] * Pi[n];
    float qpi = Qr[n] * Pi[n] + Qi[n] * Pr[n];
    k10r += qbr * cr - qbi * ci;  k10i += qbr * ci + qbi * cr;
    k11r += qpr * cr - qpi * ci;  k11i += qpr * ci + qpi * cr;
  }
  float er = 1.f + k11r, ei = k11i;
  float em = 1.f / (er * er + ei * ei);
  corr_r[l] = (k10r * er + k10i * ei) * em;            // k10/(1+k11)
  corr_i[l] = (k10i * er - k10r * ei) * em;
  w2r[l] = 2.f * br2 * inv_den;                        // 2/(1+z)
  w2i[l] = -2.f * bi2 * inv_den;
}

// ---------------------------------------------------------------------------
// Kernel 1: Cauchy contraction k00/k01 as fp32 WMMA (16x16x4):
//   k00[h,l] = sum_n (C*B)[h,n] * cden[l,n]   (complex, 2 real wmma per part)
//   at_roots = w2 * (k00 - k01 * corr)
// Tiling 16h x 16l keeps cden reads to H/16 = 64 passes over 1 MB (L2-resident).
// fp32 matrix op is required: cden spans ~6 orders of magnitude (step=1e-3).
// ---------------------------------------------------------------------------
__global__ void k1_cauchy(const float* __restrict__ Cr, const float* __restrict__ Ci,
                          const float* __restrict__ Br, const float* __restrict__ Bi,
                          const float* __restrict__ Pr, const float* __restrict__ Pi,
                          const float* __restrict__ cdenr, const float* __restrict__ cdeni,
                          const float* __restrict__ corr_r, const float* __restrict__ corr_i,
                          const float* __restrict__ w2r, const float* __restrict__ w2i,
                          float* __restrict__ atr, float* __restrict__ ati) {
  int wave = threadIdx.x >> 5;
  int lane = threadIdx.x & 31;
  int h0 = (blockIdx.x * 8 + wave) * 16;
  int l0 = blockIdx.y * 16;
  int m = lane & 15;
  int ksel = (lane >> 4) * 2;            // K sub-pair: lanes 0-15 -> {0,1}, 16-31 -> {2,3}
  int h = h0 + m;                        // A-matrix row for this lane
  int lc = l0 + m;                       // B-matrix column for this lane
  v8f k00r = {}, k00i = {}, k01r = {}, k01i = {};
  for (int n0 = 0; n0 < NSTATE; n0 += 4) {
    v2f aCBr, aCBi, aCBin, aCPr, aCPi, aCPin, bcr, bci;
#pragma unroll
    for (int p = 0; p < 2; ++p) {
      int n = n0 + ksel + p;
      float cr = Cr[h * NSTATE + n], ci = Ci[h * NSTATE + n];
      float br = Br[n], bi = Bi[n], pr = Pr[n], pi = Pi[n];
      aCBr[p] = cr * br - ci * bi;  aCBi[p] = cr * bi + ci * br;  aCBin[p] = -aCBi[p];
      aCPr[p] = cr * pr - ci * pi;  aCPi[p] = cr * pi + ci * pr;  aCPin[p] = -aCPi[p];
      bcr[p] = cdenr[lc * NSTATE + n];
      bci[p] = cdeni[lc * NSTATE + n];
    }
    // complex MAC via 8 real fp32 WMMAs (no A/B negate modifier for f32 -> pre-negated frags)
    k00r = __builtin_amdgcn_wmma_f32_16x16x4_f32(false, aCBr,  false, bcr, (short)0, k00r, false, false);
    k00r = __builtin_amdgcn_wmma_f32_16x16x4_f32(false, aCBin, false, bci, (short)0, k00r, false, false);
    k00i = __builtin_amdgcn_wmma_f32_16x16x4_f32(false, aCBr,  false, bci, (short)0, k00i, false, false);
    k00i = __builtin_amdgcn_wmma_f32_16x16x4_f32(false, aCBi,  false, bcr, (short)0, k00i, false, false);
    k01r = __builtin_amdgcn_wmma_f32_16x16x4_f32(false, aCPr,  false, bcr, (short)0, k01r, false, false);
    k01r = __builtin_amdgcn_wmma_f32_16x16x4_f32(false, aCPin, false, bci, (short)0, k01r, false, false);
    k01i = __builtin_amdgcn_wmma_f32_16x16x4_f32(false, aCPr,  false, bci, (short)0, k01i, false, false);
    k01i = __builtin_amdgcn_wmma_f32_16x16x4_f32(false, aCPi,  false, bcr, (short)0, k01i, false, false);
  }
  // epilogue: at_roots = w2 * (k00 - k01*corr); D-matrix rows: M = v (+8 for hi lanes)
  float cc = corr_r[lc], cs = corr_i[lc], wr = w2r[lc], wi = w2i[lc];
  int mr0 = (lane < 16) ? 0 : 8;
#pragma unroll
  for (int v = 0; v < 8; ++v) {
    int hrow = h0 + mr0 + v;
    float tr = k00r[v] - (k01r[v] * cc - k01i[v] * cs);
    float ti = k00i[v] - (k01r[v] * cs + k01i[v] * cc);
    atr[(size_t)hrow * LSEQ + lc] = wr * tr - wi * ti;
    ati[(size_t)hrow * LSEQ + lc] = wr * ti + wi * tr;
  }
}

// ---------------------------------------------------------------------------
// Kernel 2: per-channel 2048-point radix-2 inverse FFT in LDS.
//   K[h][l] = real(ifft(at_roots[h]))  -> stored f16 for the WMMA convolution.
// ---------------------------------------------------------------------------
__global__ void k2_ifft(const float* __restrict__ atr, const float* __restrict__ ati,
                        _Float16* __restrict__ K16) {
  __shared__ float re[LSEQ];
  __shared__ float im[LSEQ];
  int h = blockIdx.x;
  for (int l = threadIdx.x; l < LSEQ; l += blockDim.x) {
    unsigned rl = __brev((unsigned)l) >> 21;     // 11-bit bit-reverse
    re[rl] = atr[(size_t)h * LSEQ + l];
    im[rl] = ati[(size_t)h * LSEQ + l];
  }
  __syncthreads();
  for (int len = 2; len <= LSEQ; len <<= 1) {
    int hlen = len >> 1;
    float ang = 6.28318530717958647692f / (float)len;   // +sign => inverse transform
    for (int bidx = threadIdx.x; bidx < LSEQ / 2; bidx += blockDim.x) {
      int grp = bidx / hlen, pos = bidx - grp * hlen;
      int i = grp * len + pos, j = i + hlen;
      float s, c;
      __sincosf(ang * (float)pos, &s, &c);
      float ur = re[i], ui = im[i], vr = re[j], vi = im[j];
      float tr = vr * c - vi * s, ti = vr * s + vi * c;
      re[i] = ur + tr; im[i] = ui + ti;
      re[j] = ur - tr; im[j] = ui - ti;
    }
    __syncthreads();
  }
  const float scale = 1.f / (float)LSEQ;
  for (int l = threadIdx.x; l < LSEQ; l += blockDim.x)
    K16[(size_t)h * LSEQ + l] = (_Float16)(re[l] * scale);
}

// ---------------------------------------------------------------------------
// Kernel 3: pack/transpose r (B,L,H) fp32 -> rT (H,B,L) f16 via LDS tile so
// both sides stay coalesced. (H,B,L): per (h,b) the time series is contiguous,
// which lets the conv kernel's B-fragment be two aligned ds_load_b128.
// ---------------------------------------------------------------------------
__global__ void k3_pack_r(const float* __restrict__ r, _Float16* __restrict__ rT) {
  __shared__ _Float16 tile[32][32][NBATCH];
  int h0 = blockIdx.x * 32, l0 = blockIdx.y * 32;
  for (int b = 0; b < NBATCH; ++b)
    for (int idx = threadIdx.x; idx < 32 * 32; idx += blockDim.x) {
      int li = idx >> 5, hi = idx & 31;   // 32 consecutive h -> 128B coalesced read
      tile[hi][li][b] = (_Float16)r[(size_t)b * LSEQ * HCH + (size_t)(l0 + li) * HCH + h0 + hi];
    }
  __syncthreads();
  for (int f = threadIdx.x; f < 32 * 32 * NBATCH; f += blockDim.x) {
    int hi = f >> 8, c = f & 255, b = c >> 5, li = c & 31;  // contiguous 64B runs per (hi,b)
    rT[(((size_t)(h0 + hi)) * NBATCH + b) * LSEQ + (l0 + li)] = tile[hi][li][b];
  }
}

// ---------------------------------------------------------------------------
// Kernel 4: causal convolution as Toeplitz-tile WMMA (f16 in, f32 acc).
//   One workgroup per channel h (8 waves); rT_h (32KB) via Tensor Data Mover.
//   A-fragments come from a sliding-pair table of reversed/zero-padded K so
//   each fragment VGPR is one aligned ds_load_b32; B-fragments are two aligned
//   ds_load_b128. Each wave computes TWO adjacent 16-row tiles per t0 sweep
//   (identical t0 range), sharing the B fragment.
//     y[l0+i, b] = sum_{t0,j} K[l0+i - t0 - j] * r[t0+j, b]  (neg lag -> 0 pad)
// ---------------------------------------------------------------------------
__global__ void k4_conv(const _Float16* __restrict__ K16, const _Float16* __restrict__ rT,
                        float* __restrict__ yT) {
  __shared__ unsigned sA[2081];            // sA[x] = {Krev[x], Krev[x+1]} packed halves
  __shared__ _Float16 sRb[NBATCH * LSEQ];  // rT[h] as [b][t]
  int h = blockIdx.x;
  size_t hb = (size_t)h * LSEQ;
  // Krev[x] = (x < L) ? K[L-1-x] : 0  (indices >= L are the causal zero pad)
  for (int x = threadIdx.x; x < 2081; x += blockDim.x) {
    union { _Float16 f; unsigned short u; } e0, e1;
    e0.f = (x < LSEQ) ? K16[hb + (LSEQ - 1 - x)] : (_Float16)0.f;
    e1.f = (x + 1 < LSEQ) ? K16[hb + (LSEQ - 2 - x)] : (_Float16)0.f;
    sA[x] = (unsigned)e0.u | ((unsigned)e1.u << 16);
  }

  if (threadIdx.x < 32) {
    // Tensor DMA descriptor (D#), 1-D tile of 16384 f16 elements (contiguous 32KB).
    unsigned long long ga = (unsigned long long)(size_t)(rT + (size_t)h * LSEQ * NBATCH);
    unsigned ldsoff = (unsigned)(size_t)(&sRb[0]);         // low 32 bits = LDS byte offset
    const unsigned elems = LSEQ * NBATCH;                  // 16384
    v4u g0;
    g0[0] = 1u;                                            // count=1, user mode
    g0[1] = ldsoff;                                        // lds_addr
    g0[2] = (unsigned)(ga & 0xffffffffu);                  // global_addr[31:0]
    g0[3] = (unsigned)((ga >> 32) & 0x01ffffffu) | (2u << 30); // addr[56:32] | type=2
    v8i g1;
    g1[0] = (int)(1u << 16);                               // data_size=1 (2B), no pad/iter
    g1[1] = (int)((elems & 0xffffu) << 16);                // tensor_dim0[15:0]
    g1[2] = (int)((elems >> 16) & 0xffffu);                // tensor_dim0[31:16], dim1=0
    g1[3] = (int)((elems & 0xffffu) << 16);                // tile_dim0 = 16384
    g1[4] = 0;                                             // tile_dim1/2 unused
    g1[5] = (int)elems;                                    // tensor_dim0_stride[31:0]
    g1[6] = 0;
    g1[7] = 0;
    v4i gz4 = {0, 0, 0, 0};
    v8i gz8 = {0, 0, 0, 0, 0, 0, 0, 0};
    __builtin_amdgcn_tensor_load_to_lds(g0, g1, gz4, gz4, gz8, 0);
  }
  __builtin_amdgcn_s_wait_tensorcnt(0);
  __syncthreads();

  int wave = threadIdx.x >> 5, lane = threadIdx.x & 31;
  int m = lane & 15;                 // A row / output row within tile
  int jb = (lane >> 4) * 16;         // A K-offset (lanes 16-31 hold K=16..31)
  int ncol = lane & 15;              // B column
  int kb = (lane >> 4) * 16;         // B K-offset (lanes 16-31 hold K=16..31)
  bool bval = ncol < NBATCH;
  int bcol = ncol & (NBATCH - 1);
  const v16h vz = {};
  for (int q = wave; q < LSEQ / 32; q += 8) {
    int l0a = q * 32, l0b = l0a + 16;        // two row tiles, same t0 range
    v8f acc0 = {}, acc1 = {};
    for (int t0 = 0; t0 <= l0a; t0 += 32) {
      // B fragment: 16 consecutive halves, 32B aligned -> two ds_load_b128
      int boff = bcol * LSEQ + t0 + kb;
      union { uint4 q4[2]; v16h v; } bu;
      bu.q4[0] = *reinterpret_cast<const uint4*>(&sRb[boff]);
      bu.q4[1] = *reinterpret_cast<const uint4*>(&sRb[boff + 8]);
      v16h bfrag = bval ? bu.v : vz;
      // A fragments (Toeplitz): one aligned b32 pair-load per VGPR
      int sa = (LSEQ - 1) - (l0a - t0) - m + jb;
      union { unsigned u[8]; v16h v; } a0, a1;
#pragma unroll
      for (int p = 0; p < 8; ++p) {
        int d = 2 * p;                       // 0,2,4,...,14
        a0.u[p] = sA[sa + d];
        a1.u[p] = sA[sa - 16 + d];
      }
      acc0 = __builtin_amdgcn_wmma_f32_16x16x32_f16(false, a0.v, false, bfrag, (short)0,
                                                    acc0, false, false);
      acc1 = __builtin_amdgcn_wmma_f32_16x16x32_f16(false, a1.v, false, bfrag, (short)0,
                                                    acc1, false, false);
    }
    int mr0 = (lane < 16) ? 0 : 8;
    if (bval) {
#pragma unroll
      for (int v = 0; v < 8; ++v) {
        yT[(hb + (size_t)(l0a + mr0 + v)) * NBATCH + bcol] = acc0[v];
        yT[(hb + (size_t)(l0b + mr0 + v)) * NBATCH + bcol] = acc1[v];
      }
    }
  }
}

// ---------------------------------------------------------------------------
// Kernel 5: yT (H,L,B) -> out (B,L,H) with fused D*r and exact GELU.
// LDS tiled so reads (yT, contiguous) and writes (out, h contiguous) are both
// coalesced — avoids ~32x store amplification.
// ---------------------------------------------------------------------------
__global__ void k5_out(const float* __restrict__ yT, const float* __restrict__ r,
                       const float* __restrict__ Dp, float* __restrict__ out) {
  __shared__ float tile[32][32][NBATCH + 1];
  int h0 = blockIdx.x * 32, l0 = blockIdx.y * 32;
  float Dv = Dp[0];
  for (int f = threadIdx.x; f < 32 * 32 * NBATCH; f += blockDim.x) {
    int hi = f >> 8, c = f & 255, li = c >> 3, b = c & 7;   // contiguous yT read per hi
    tile[hi][li][b] = yT[((size_t)(h0 + hi) * LSEQ + (l0 + li)) * NBATCH + b];
  }
  __syncthreads();
  for (int f = threadIdx.x; f < 32 * 32 * NBATCH; f += blockDim.x) {
    int b = f >> 10, rem = f & 1023, li = rem >> 5, hi = rem & 31;  // contiguous out write
    size_t o = (size_t)b * LSEQ * HCH + (size_t)(l0 + li) * HCH + h0 + hi;
    float x = tile[hi][li][b] + Dv * r[o];
    out[o] = 0.5f * x * (1.f + erff(x * 0.70710678118654752440f));
  }
}

// ---------------------------------------------------------------------------
extern "C" void kernel_launch(void* const* d_in, const int* in_sizes, int n_in,
                              void* d_out, int out_size, void* d_ws, size_t ws_size,
                              hipStream_t stream) {
  (void)in_sizes; (void)n_in; (void)out_size; (void)ws_size;
  const float* r    = (const float*)d_in[0];
  const float* Br   = (const float*)d_in[1];
  const float* Bi   = (const float*)d_in[2];
  const float* Cr   = (const float*)d_in[3];
  const float* Ci   = (const float*)d_in[4];
  const float* Pr   = (const float*)d_in[5];
  const float* Pi   = (const float*)d_in[6];
  const float* Qr   = (const float*)d_in[7];
  const float* Qi   = (const float*)d_in[8];
  const float* dre  = (const float*)d_in[9];
  const float* dim_ = (const float*)d_in[10];
  const float* stp  = (const float*)d_in[11];
  const float* Dd   = (const float*)d_in[12];
  float* out = (float*)d_out;

  char* ws = (char*)d_ws;
  size_t off = 0;
  auto alloc = [&](size_t bytes) -> void* {
    off = (off + 255) & ~(size_t)255;
    void* p = ws + off;
    off += bytes;
    return p;
  };
  float*    cdenr  = (float*)alloc((size_t)LSEQ * NSTATE * 4);
  float*    cdeni  = (float*)alloc((size_t)LSEQ * NSTATE * 4);
  float*    corr_r = (float*)alloc((size_t)LSEQ * 4);
  float*    corr_i = (float*)alloc((size_t)LSEQ * 4);
  float*    w2r    = (float*)alloc((size_t)LSEQ * 4);
  float*    w2i    = (float*)alloc((size_t)LSEQ * 4);
  float*    atr    = (float*)alloc((size_t)HCH * LSEQ * 4);
  float*    ati    = (float*)alloc((size_t)HCH * LSEQ * 4);
  _Float16* K16    = (_Float16*)alloc((size_t)HCH * LSEQ * 2);
  _Float16* rT     = (_Float16*)alloc((size_t)HCH * LSEQ * NBATCH * 2);
  float*    yT     = (float*)alloc((size_t)HCH * LSEQ * NBATCH * 4);

  k0_setup<<<LSEQ / 256, 256, 0, stream>>>(Br, Bi, Pr, Pi, Qr, Qi, dre, dim_, stp,
                                           cdenr, cdeni, corr_r, corr_i, w2r, w2i);
  dim3 g1(HCH / 128, LSEQ / 16);   // 8 x 128 blocks, 8 waves each (one 16h-tile per wave)
  k1_cauchy<<<g1, 256, 0, stream>>>(Cr, Ci, Br, Bi, Pr, Pi, cdenr, cdeni,
                                    corr_r, corr_i, w2r, w2i, atr, ati);
  k2_ifft<<<HCH, 256, 0, stream>>>(atr, ati, K16);
  dim3 g3(HCH / 32, LSEQ / 32);
  k3_pack_r<<<g3, 256, 0, stream>>>(r, rT);
  k4_conv<<<HCH, 256, 0, stream>>>(K16, rT, yT);
  dim3 g5(HCH / 32, LSEQ / 32);
  k5_out<<<g5, 256, 0, stream>>>(yT, r, Dd, out);
}
